// CrossAttention_66589172957767
// MI455X (gfx1250) — compile-verified, matrix operand order checked by reading, not run on hardware
//
#include <hip/hip_runtime.h>
#include <hip/hip_bf16.h>
#include <math.h>

typedef __bf16 bf16_t;
typedef __attribute__((ext_vector_type(16))) __bf16 v16bf;
typedef __attribute__((ext_vector_type(8)))  __bf16 v8bf;
typedef __attribute__((ext_vector_type(4)))  __bf16 v4bf;
typedef __attribute__((ext_vector_type(8)))  float  v8f;

#define BQ   4
#define SQ   2048
#define DM   1024
#define NH   16
#define DH   64
#define HEADS_ELEMS (4*16*2048*64)   // 8388608 elements per q/k/v buffer

#define WMMA_BF16(A, Bm, C) \
  __builtin_amdgcn_wmma_f32_16x16x32_bf16(false, (A), false, (Bm), (short)0, (C), false, false)

// ---------------------------------------------------------------------------
// Projection GEMM: out = X[8192,1024] @ W[1024,1024] + bias, emitted as bf16
// head-split [b,h,s,64] (transpose_store=0) or [b,h,64,s] (transpose_store=1).
// Block tile 128(M) x 128(N), 8 waves each owning 32x64; K-step 32.
// 8 WMMA per wave per K-step, B-fragments reused across both row-groups.
// ---------------------------------------------------------------------------
__global__ __launch_bounds__(256)
void proj_kernel(const float* __restrict__ X, const float* __restrict__ W,
                 const float* __restrict__ bias, bf16_t* __restrict__ out,
                 int transpose_store) {
    __shared__ bf16_t At[128 * 32];   // [row][k]   row-major, k contiguous
    __shared__ bf16_t Bt[128 * 32];   // [n][k]     transposed, k contiguous

    const int tid   = threadIdx.x;
    const int wave  = tid >> 5;
    const int lane  = tid & 31;
    const int ln    = lane & 15;
    const int hi    = lane >> 4;
    const int kb8   = hi * 8;

    const int m0 = blockIdx.y * 128;
    const int n0 = blockIdx.x * 128;
    const int waveM = wave >> 1;      // 0..3  (32-row strip)
    const int waveN = wave & 1;       // 0..1  (64-col strip)

    v8f acc[2][4];
    #pragma unroll
    for (int rg = 0; rg < 2; ++rg)
        #pragma unroll
        for (int c = 0; c < 4; ++c) acc[rg][c] = v8f{};

    for (int kk = 0; kk < DM / 32; ++kk) {
        const int kbase = kk * 32;

        // ---- stage A: 128x32 f32 -> bf16, 1024 float4 segments, 4/thread --
        #pragma unroll
        for (int i = 0; i < 4; ++i) {
            const int idx = tid + 256 * i;
            const int row = idx >> 3;
            const int c4  = idx & 7;
            const float4 f = *(const float4*)(X + (size_t)(m0 + row) * DM + kbase + c4 * 4);
            v4bf h = { (bf16_t)f.x, (bf16_t)f.y, (bf16_t)f.z, (bf16_t)f.w };
            *(v4bf*)&At[row * 32 + c4 * 4] = h;
        }
        // ---- stage B transposed: 32x128 f32 -> bf16, 1024 float4 segs -----
        #pragma unroll
        for (int i = 0; i < 4; ++i) {
            const int idx = tid + 256 * i;
            const int row = idx >> 5;       // k within tile (0..31)
            const int c4  = idx & 31;       // n/4 within tile
            const float4 f = *(const float4*)(W + (size_t)(kbase + row) * DM + n0 + c4 * 4);
            Bt[(c4 * 4 + 0) * 32 + row] = (bf16_t)f.x;
            Bt[(c4 * 4 + 1) * 32 + row] = (bf16_t)f.y;
            Bt[(c4 * 4 + 2) * 32 + row] = (bf16_t)f.z;
            Bt[(c4 * 4 + 3) * 32 + row] = (bf16_t)f.w;
        }
        __syncthreads();

        // ---- WMMA: 2 A-frags x 4 B-frags = 8 WMMA; B reused across rows ---
        v16bf afrag[2];
        #pragma unroll
        for (int rg = 0; rg < 2; ++rg) {
            const bf16_t* ap = &At[(waveM * 32 + rg * 16 + ln) * 32 + kb8];
            *((v8bf*)&afrag[rg])     = *(const v8bf*)(ap);
            *((v8bf*)&afrag[rg] + 1) = *(const v8bf*)(ap + 16);
        }
        #pragma unroll
        for (int c = 0; c < 4; ++c) {
            v16bf bfrag;
            const bf16_t* bp = &Bt[(waveN * 64 + c * 16 + ln) * 32 + kb8];
            *((v8bf*)&bfrag)     = *(const v8bf*)(bp);
            *((v8bf*)&bfrag + 1) = *(const v8bf*)(bp + 16);
            acc[0][c] = WMMA_BF16(afrag[0], bfrag, acc[0][c]);
            acc[1][c] = WMMA_BF16(afrag[1], bfrag, acc[1][c]);
        }
        __syncthreads();
    }

    // ---- epilogue: bias add + bf16 store, head-split layout ---------------
    #pragma unroll
    for (int rg = 0; rg < 2; ++rg) {
        #pragma unroll
        for (int c = 0; c < 4; ++c) {
            const int n  = n0 + waveN * 64 + c * 16 + ln;
            const float bv = bias[n];
            const int h = n >> 6;
            const int d = n & 63;
            if (!transpose_store) {
                #pragma unroll
                for (int r = 0; r < 8; ++r) {
                    const int m = m0 + waveM * 32 + rg * 16 + r + hi * 8;
                    const int b = m >> 11;
                    const int s = m & 2047;
                    out[(((size_t)(b * NH + h) * SQ) + s) * DH + d] =
                        (bf16_t)(acc[rg][c][r] + bv);
                }
            } else {
                // 8 rows a lane owns are consecutive s -> one b128 store
                const int mbase = m0 + waveM * 32 + rg * 16 + hi * 8;
                const int b = mbase >> 11;
                const int s = mbase & 2047;
                v8bf pk;
                #pragma unroll
                for (int r = 0; r < 8; ++r) pk[r] = (bf16_t)(acc[rg][c][r] + bv);
                *(v8bf*)&out[(((size_t)(b * NH + h) * DH) + d) * SQ + s] = pk;
            }
        }
    }
}

// ---------------------------------------------------------------------------
// Flash attention: per (b,h,qtile64). 4 waves x 16 query rows. K/V^T chunks
// of 32 keys, double-buffered in LDS via global_load_async_to_lds_b128.
// ---------------------------------------------------------------------------
__global__ __launch_bounds__(128)
void attn_kernel(const bf16_t* __restrict__ qb, const bf16_t* __restrict__ kb,
                 const bf16_t* __restrict__ vtb, float* __restrict__ out) {
    __shared__ bf16_t Kt[2][32 * 64];   // [pos][d]  (d contiguous)
    __shared__ bf16_t Vt[2][64 * 32];   // [d][pos]  (pos contiguous)
    __shared__ bf16_t Pt[4][16 * 32];   // per-wave P scratch [row][keycol]

    const int tid  = threadIdx.x;
    const int wave = tid >> 5;
    const int lane = tid & 31;
    const int ln   = lane & 15;
    const int hi   = lane >> 4;
    const int kb8  = hi * 8;

    const int bh = blockIdx.z * NH + blockIdx.y;
    const int q0 = blockIdx.x * 64 + wave * 16;

    // Q fragments (A layout), resident in registers for the whole loop.
    v16bf qf[2];
    {
        const bf16_t* qp = qb + ((size_t)bh * SQ + q0 + ln) * DH;
        *((v8bf*)&qf[0])     = *(const v8bf*)(qp + kb8);
        *((v8bf*)&qf[0] + 1) = *(const v8bf*)(qp + kb8 + 16);
        *((v8bf*)&qf[1])     = *(const v8bf*)(qp + 32 + kb8);
        *((v8bf*)&qf[1] + 1) = *(const v8bf*)(qp + 32 + kb8 + 16);
    }

    const unsigned long long ksrc = (unsigned long long)(uintptr_t)(kb  + (size_t)bh * SQ * DH);
    const unsigned long long vsrc = (unsigned long long)(uintptr_t)(vtb + (size_t)bh * DH * SQ);

    auto stage = [&](int j, int buf) {
        const unsigned ktbase = (unsigned)(uintptr_t)&Kt[buf][0];
        const unsigned vtbase = (unsigned)(uintptr_t)&Vt[buf][0];
        #pragma unroll
        for (int i = 0; i < 2; ++i) {                       // K chunk: 256 x 16B
            const int seg = tid * 2 + i;
            const int pos = seg >> 3;
            const int d8  = (seg & 7) * 8;
            const unsigned voff = (unsigned)(((j * 32 + pos) * DH + d8) * 2);
            const unsigned ldst = ktbase + (unsigned)((pos * DH + d8) * 2);
            asm volatile("global_load_async_to_lds_b128 %0, %1, %2"
                         :: "v"(ldst), "v"(voff), "s"(ksrc) : "memory");
        }
        #pragma unroll
        for (int i = 0; i < 2; ++i) {                       // V^T chunk: 256 x 16B
            const int seg = tid * 2 + i;
            const int d   = seg >> 2;
            const int p8  = (seg & 3) * 8;
            const unsigned voff = (unsigned)((d * SQ + j * 32 + p8) * 2);
            const unsigned ldst = vtbase + (unsigned)((d * 32 + p8) * 2);
            asm volatile("global_load_async_to_lds_b128 %0, %1, %2"
                         :: "v"(ldst), "v"(voff), "s"(vsrc) : "memory");
        }
    };

    float m[8], l[8];
    v8f ctx[4] = {v8f{}, v8f{}, v8f{}, v8f{}};
    #pragma unroll
    for (int r = 0; r < 8; ++r) { m[r] = -1e30f; l[r] = 0.0f; }

    const float scale = 0.125f;           // 1/sqrt(64)
    const float LOG2E = 1.44269504f;

    stage(0, 0);

    for (int j = 0; j < SQ / 32; ++j) {
        asm volatile("s_wait_asynccnt 0" ::: "memory");
        __syncthreads();
        if (j + 1 < SQ / 32) stage(j + 1, (j + 1) & 1);

        const bf16_t* K = &Kt[j & 1][0];
        const bf16_t* V = &Vt[j & 1][0];

        // ---- scores: 2 col-tiles x (Kdim 64 = 2 WMMA) ----
        v8f sc[2];
        #pragma unroll
        for (int t = 0; t < 2; ++t) {
            v8f s = v8f{};
            #pragma unroll
            for (int step = 0; step < 2; ++step) {
                v16bf bfrag;
                const bf16_t* bp = K + (t * 16 + ln) * DH + step * 32 + kb8;
                *((v8bf*)&bfrag)     = *(const v8bf*)(bp);
                *((v8bf*)&bfrag + 1) = *(const v8bf*)(bp + 16);
                s = WMMA_BF16(qf[step], bfrag, s);
            }
            sc[t] = s;
        }

        // ---- online softmax over the 32 new key columns ----
        float al[8];
        #pragma unroll
        for (int r = 0; r < 8; ++r) {
            float a  = sc[0][r] * scale;
            float b2 = sc[1][r] * scale;
            float mx = fmaxf(a, b2);
            mx = fmaxf(mx, __shfl_xor(mx, 1, 32));
            mx = fmaxf(mx, __shfl_xor(mx, 2, 32));
            mx = fmaxf(mx, __shfl_xor(mx, 4, 32));
            mx = fmaxf(mx, __shfl_xor(mx, 8, 32));
            const float mnew = fmaxf(m[r], mx);
            al[r] = exp2f((m[r] - mnew) * LOG2E);
            const float p0 = exp2f((a  - mnew) * LOG2E);
            const float p1 = exp2f((b2 - mnew) * LOG2E);
            float rs = p0 + p1;
            rs += __shfl_xor(rs, 1, 32);
            rs += __shfl_xor(rs, 2, 32);
            rs += __shfl_xor(rs, 4, 32);
            rs += __shfl_xor(rs, 8, 32);
            l[r] = l[r] * al[r] + rs;
            m[r] = mnew;
            // C-layout -> LDS (per-wave region), bf16
            Pt[wave][(r + hi * 8) * 32 + ln]      = (bf16_t)p0;
            Pt[wave][(r + hi * 8) * 32 + 16 + ln] = (bf16_t)p1;
        }

        // ---- rescale running context ----
        #pragma unroll
        for (int c = 0; c < 4; ++c)
            #pragma unroll
            for (int r = 0; r < 8; ++r)
                ctx[c][r] *= al[r];

        // ---- P as A-fragment (wave-local LDS round trip, DS in-order) ----
        v16bf pf;
        {
            const bf16_t* pp = &Pt[wave][ln * 32 + kb8];
            *((v8bf*)&pf)     = *(const v8bf*)(pp);
            *((v8bf*)&pf + 1) = *(const v8bf*)(pp + 16);
        }

        // ---- PV: 4 d-tiles, B-frag from V^T (pos contiguous) ----
        #pragma unroll
        for (int c = 0; c < 4; ++c) {
            v16bf bfrag;
            const bf16_t* bp = V + (c * 16 + ln) * 32 + kb8;
            *((v8bf*)&bfrag)     = *(const v8bf*)(bp);
            *((v8bf*)&bfrag + 1) = *(const v8bf*)(bp + 16);
            ctx[c] = WMMA_BF16(pf, bfrag, ctx[c]);
        }
    }

    // ---- epilogue: divide by row sums, store f32 [b,s,1024] ----
    #pragma unroll
    for (int c = 0; c < 4; ++c) {
        const int d = c * 16 + ln;
        #pragma unroll
        for (int r = 0; r < 8; ++r) {
            const int row = q0 + r + hi * 8;
            const float v = ctx[c][r] / l[r];
            out[((size_t)blockIdx.z * SQ + row) * DM + blockIdx.y * DH + d] = v;
        }
    }
}

// ---------------------------------------------------------------------------
extern "C" void kernel_launch(void* const* d_in, const int* in_sizes, int n_in,
                              void* d_out, int out_size, void* d_ws, size_t ws_size,
                              hipStream_t stream) {
    const float* query  = (const float*)d_in[0];
    const float* key_in = (const float*)d_in[1];
    const float* value  = (const float*)d_in[2];
    const float* Wq     = (const float*)d_in[3];
    const float* bq     = (const float*)d_in[4];
    const float* Wk     = (const float*)d_in[5];
    const float* bk     = (const float*)d_in[6];
    const float* Wv     = (const float*)d_in[7];
    const float* bv     = (const float*)d_in[8];
    float* out = (float*)d_out;

    bf16_t* qb  = (bf16_t*)d_ws;              // [b,h,s,64] bf16
    bf16_t* kb  = qb + HEADS_ELEMS;           // [b,h,s,64] bf16
    bf16_t* vtb = kb + HEADS_ELEMS;           // [b,h,64,s] bf16

    dim3 pg(DM / 128, (BQ * SQ) / 128);       // 8 x 64
    dim3 pb(256);
    proj_kernel<<<pg, pb, 0, stream>>>(query,  Wq, bq, qb,  0);
    proj_kernel<<<pg, pb, 0, stream>>>(key_in, Wk, bk, kb,  0);
    proj_kernel<<<pg, pb, 0, stream>>>(value,  Wv, bv, vtb, 1);

    dim3 ag(SQ / 64, NH, BQ);                 // 32 x 16 x 4
    dim3 ab(128);
    attn_kernel<<<ag, ab, 0, stream>>>(qb, kb, vtb, out);
}
